// Encoder_37821482008624
// MI455X (gfx1250) — compile-verified, matrix operand order checked by reading, not run on hardware
//
#include <hip/hip_runtime.h>
#include <hip/hip_bf16.h>
#include <math.h>

typedef __attribute__((ext_vector_type(16))) _Float16 v16h;
typedef __attribute__((ext_vector_type(8)))  _Float16 v8h;
typedef __attribute__((ext_vector_type(8)))  float    v8f;

static constexpr int BATCH = 16384;
static constexpr int TT    = 20;
static constexpr int FF    = 80;
static constexpr int HH    = 256;   // hidden
static constexpr int ZD    = 32;
static constexpr int GN    = 4 * HH;      // 1024 gate columns
static constexpr size_t OUTS = (size_t)BATCH * ZD;  // one output tensor

// ---------------------------------------------------------------------------
// Weight repack: f32 (K x 1024) row-major  ->  f16 WMMA-B tiles.
// Tile (nt, kt) covers N=[nt*16,nt*16+16), K=[kt*32,kt*32+32), stored as
// 512 halfs: offset = (nt*KT+kt)*512 + lane*16 + e, with the 16x16x32 f16
// B-operand lane mapping: lane = n%16 + 16*khalf;
//   e<8  -> k = kt*32 + khalf*8 + e
//   e>=8 -> k = kt*32 + 16 + khalf*8 + (e-8)
// Zero-padded past Kact so F=80 -> 96 and F+ZD=112 -> 128 need no edge code.
// ---------------------------------------------------------------------------
__global__ void pack_weights(const float* __restrict__ src, _Float16* __restrict__ dst,
                             int Kact, int KT) {
    int idx = blockIdx.x * 256 + threadIdx.x;
    int total = KT * 64 * 512;
    if (idx >= total) return;
    int e     = idx & 15;
    int lane  = (idx >> 4) & 31;
    int tile  = idx >> 9;
    int kt    = tile % KT;
    int nt    = tile / KT;
    int n     = nt * 16 + (lane & 15);
    int khalf = lane >> 4;
    int k     = kt * 32 + ((e < 8) ? (khalf * 8 + e) : (16 + khalf * 8 + (e - 8)));
    float v = (k < Kact) ? src[(size_t)k * GN + n] : 0.0f;
    dst[idx] = (_Float16)v;
}

// ---------------------------------------------------------------------------
struct __align__(32) SMem {
    _Float16 in[16 * 128];    // staged x_t (|| z2_sample) padded, f16 row-major
    _Float16 h1[16 * 256];    // layer-1 hidden, f16 row-major
    _Float16 h2[16 * 256];
    _Float16 z2s[16 * 32];    // z2_sample (f16) feeding the z1 branch
    float    c1[16 * 256];    // cell states f32
    float    c2[16 * 256];
    float    z[16 * 1024];    // raw gate pre-activations
};

__device__ __forceinline__ float sigf(float x) { return 1.0f / (1.0f + __expf(-x)); }

// A-fragment (16x32, f16) from row-major LDS: two contiguous 16B loads/lane.
__device__ __forceinline__ v16h load_a(const _Float16* base, int ld, int kt, int m, int khalf) {
    const _Float16* r = base + m * ld + kt * 32 + khalf * 8;
    v8h lo = *(const v8h*)(r);
    v8h hi = *(const v8h*)(r + 16);
    return __builtin_shufflevector(lo, hi, 0,1,2,3,4,5,6,7,8,9,10,11,12,13,14,15);
}

// z[16 x 1024] = sx @ W + sh @ U + b.  Wave w owns N columns [w*128,(w+1)*128).
template <int KT0>
__device__ void gates(const _Float16* __restrict__ sx, const _Float16* __restrict__ sh,
                      const _Float16* __restrict__ Wp, const _Float16* __restrict__ Up,
                      const float* __restrict__ bias, float* __restrict__ sz) {
    const int lane  = threadIdx.x & 31;
    const int wave  = threadIdx.x >> 5;
    const int m     = lane & 15;
    const int khalf = lane >> 4;

    v16h ax[KT0];
#pragma unroll
    for (int kt = 0; kt < KT0; ++kt) ax[kt] = load_a(sx, KT0 * 32, kt, m, khalf);
    v16h ah[8];
#pragma unroll
    for (int kt = 0; kt < 8; ++kt) ah[kt] = load_a(sh, 256, kt, m, khalf);

#pragma unroll
    for (int ntl = 0; ntl < 8; ++ntl) {
        const int nt = wave * 8 + ntl;
        const float bv = bias[nt * 16 + m];
        v8f acc = {bv, bv, bv, bv, bv, bv, bv, bv};

        const v16h* wp = (const v16h*)Wp + (size_t)(nt * KT0) * 32 + lane;
#pragma unroll
        for (int kt = 0; kt < KT0; ++kt)
            acc = __builtin_amdgcn_wmma_f32_16x16x32_f16(false, ax[kt], false, wp[kt * 32],
                                                         (short)0, acc, false, false);

        const v16h* up = (const v16h*)Up + (size_t)(nt * 8) * 32 + lane;
        if (ntl < 7)  // warm L1/L2 for the next column group's U tiles
            __builtin_prefetch((const void*)((const v16h*)Up + (size_t)((nt + 1) * 8) * 32 + lane), 0, 1);
#pragma unroll
        for (int kt = 0; kt < 8; ++kt)
            acc = __builtin_amdgcn_wmma_f32_16x16x32_f16(false, ah[kt], false, up[kt * 32],
                                                         (short)0, acc, false, false);

        const int n  = nt * 16 + m;
        const int m0 = khalf * 8;
#pragma unroll
        for (int r = 0; r < 8; ++r) sz[(m0 + r) * 1024 + n] = acc[r];
    }
}

// LSTM pointwise: c' = sig(f)*c + sig(i)*tanh(g); h' = sig(o)*tanh(c')
__device__ void combine(const float* __restrict__ sz, float* __restrict__ sc,
                        _Float16* __restrict__ sh) {
    const int j = threadIdx.x;  // 0..255 = hidden index
#pragma unroll 4
    for (int m = 0; m < 16; ++m) {
        float zi = sz[m * 1024 + j];
        float zf = sz[m * 1024 + 256 + j];
        float zg = sz[m * 1024 + 512 + j];
        float zo = sz[m * 1024 + 768 + j];
        float c  = sc[m * 256 + j];
        float cn = sigf(zf) * c + sigf(zi) * tanhf(zg);
        sc[m * 256 + j] = cn;
        sh[m * 256 + j] = (_Float16)(sigf(zo) * tanhf(cn));
    }
}

__device__ void stage(SMem* sm, const float* __restrict__ x, int b0, int t, int LD, bool is_z1) {
    for (int idx = threadIdx.x; idx < 16 * LD; idx += 256) {
        int m = idx / LD, f = idx - m * LD;
        _Float16 v = (_Float16)0.0f;
        if (f < FF)                      v = (_Float16)x[((size_t)(b0 + m) * TT + t) * FF + f];
        else if (is_z1 && f < FF + ZD)   v = sm->z2s[m * ZD + (f - FF)];
        sm->in[idx] = v;
    }
}

__device__ void heads(SMem* sm, const float* __restrict__ muW, const float* __restrict__ mub,
                      const float* __restrict__ lvW, const float* __restrict__ lvb,
                      const float* __restrict__ eps, float* __restrict__ omu,
                      float* __restrict__ olv, float* __restrict__ osmp, int b0, bool store_z2s) {
    for (int e = threadIdx.x; e < 16 * ZD; e += 256) {
        int m = e >> 5, zd = e & 31;
        float smu = mub[zd], slv = lvb[zd];
#pragma unroll 8
        for (int k = 0; k < 256; ++k) {
            float c = sm->c1[m * 256 + k];
            smu += c * muW[(size_t)k * ZD + zd];
            slv += c * lvW[(size_t)k * ZD + zd];
        }
#pragma unroll 8
        for (int k = 0; k < 256; ++k) {
            float c = sm->c2[m * 256 + k];
            smu += c * muW[(size_t)(256 + k) * ZD + zd];
            slv += c * lvW[(size_t)(256 + k) * ZD + zd];
        }
        size_t g = (size_t)(b0 + m) * ZD + zd;
        float s = eps[g] * __expf(0.5f * slv) + smu;
        omu[g] = smu; olv[g] = slv; osmp[g] = s;
        if (store_z2s) sm->z2s[m * ZD + zd] = (_Float16)s;
    }
}

template <int KT0>
__device__ void run_branch(SMem* sm, const float* x,
                           const _Float16* W1p, const _Float16* U1p, const float* b1,
                           const _Float16* W2p, const _Float16* U2p, const float* b2,
                           const float* muW, const float* mub, const float* lvW, const float* lvb,
                           const float* eps, float* omu, float* olv, float* osmp, bool is_z1) {
    const int b0 = blockIdx.x * 16;
    for (int i = threadIdx.x; i < 16 * 256; i += 256) {
        sm->h1[i] = (_Float16)0.0f; sm->h2[i] = (_Float16)0.0f;
        sm->c1[i] = 0.0f;           sm->c2[i] = 0.0f;
    }
    __syncthreads();
    for (int t = 0; t < TT; ++t) {
        stage(sm, x, b0, t, KT0 * 32, is_z1);                 __syncthreads();
        gates<KT0>(sm->in, sm->h1, W1p, U1p, b1, sm->z);      __syncthreads();
        combine(sm->z, sm->c1, sm->h1);                       __syncthreads();
        gates<8>(sm->h1, sm->h2, W2p, U2p, b2, sm->z);        __syncthreads();
        combine(sm->z, sm->c2, sm->h2);                       __syncthreads();
    }
    heads(sm, muW, mub, lvW, lvb, eps, omu, olv, osmp, b0, !is_z1);
    __syncthreads();
}

struct EncArgs {
    const float *x, *eps1, *eps2;
    const float *z2b1, *z2b2, *z1b1, *z1b2;
    const float *mu2W, *mu2b, *lv2W, *lv2b, *mu1W, *mu1b, *lv1W, *lv1b;
    const _Float16 *z2W1p, *z2U1p, *z2W2p, *z2U2p;
    const _Float16 *z1W1p, *z1U1p, *z1W2p, *z1U2p;
    float* out;
};

__global__ __launch_bounds__(256) void encoder_kernel(EncArgs a) {
    __shared__ SMem sm;
    // z2 branch (input F=80 -> 3 K-tiles) writes out[3..5] and caches z2_sample
    run_branch<3>(&sm, a.x, a.z2W1p, a.z2U1p, a.z2b1, a.z2W2p, a.z2U2p, a.z2b2,
                  a.mu2W, a.mu2b, a.lv2W, a.lv2b, a.eps2,
                  a.out + 3 * OUTS, a.out + 4 * OUTS, a.out + 5 * OUTS, false);
    // z1 branch (input F+ZD=112 -> 4 K-tiles) writes out[0..2]
    run_branch<4>(&sm, a.x, a.z1W1p, a.z1U1p, a.z1b1, a.z1W2p, a.z1U2p, a.z1b2,
                  a.mu1W, a.mu1b, a.lv1W, a.lv1b, a.eps1,
                  a.out + 0 * OUTS, a.out + 1 * OUTS, a.out + 2 * OUTS, true);
}

extern "C" void kernel_launch(void* const* d_in, const int* in_sizes, int n_in,
                              void* d_out, int out_size, void* d_ws, size_t ws_size,
                              hipStream_t stream) {
    (void)in_sizes; (void)n_in; (void)out_size; (void)ws_size;
    const float* x     = (const float*)d_in[0];
    const float* eps1  = (const float*)d_in[1];
    const float* eps2  = (const float*)d_in[2];
    const float* z2_W1 = (const float*)d_in[3];
    const float* z2_U1 = (const float*)d_in[4];
    const float* z2_b1 = (const float*)d_in[5];
    const float* z2_W2 = (const float*)d_in[6];
    const float* z2_U2 = (const float*)d_in[7];
    const float* z2_b2 = (const float*)d_in[8];
    const float* z1_W1 = (const float*)d_in[9];
    const float* z1_U1 = (const float*)d_in[10];
    const float* z1_b1 = (const float*)d_in[11];
    const float* z1_W2 = (const float*)d_in[12];
    const float* z1_U2 = (const float*)d_in[13];
    const float* z1_b2 = (const float*)d_in[14];
    const float* mu2W  = (const float*)d_in[15];
    const float* mu2b  = (const float*)d_in[16];
    const float* lv2W  = (const float*)d_in[17];
    const float* lv2b  = (const float*)d_in[18];
    const float* mu1W  = (const float*)d_in[19];
    const float* mu1b  = (const float*)d_in[20];
    const float* lv1W  = (const float*)d_in[21];
    const float* lv1b  = (const float*)d_in[22];

    // Workspace layout (f16 halfs); each region is a multiple of 512 halfs.
    _Float16* ws = (_Float16*)d_ws;
    _Float16* p_z2W1 = ws;                 // 3*64*512 =  98304
    _Float16* p_z2U1 = p_z2W1 +  98304;    // 8*64*512 = 262144
    _Float16* p_z2W2 = p_z2U1 + 262144;
    _Float16* p_z2U2 = p_z2W2 + 262144;
    _Float16* p_z1W1 = p_z2U2 + 262144;    // 4*64*512 = 131072
    _Float16* p_z1U1 = p_z1W1 + 131072;
    _Float16* p_z1W2 = p_z1U1 + 262144;
    _Float16* p_z1U2 = p_z1W2 + 262144;    // end: 1,802,240 halfs (~3.6 MB)

    pack_weights<<< 384, 256, 0, stream>>>(z2_W1, p_z2W1,  80, 3);
    pack_weights<<<1024, 256, 0, stream>>>(z2_U1, p_z2U1, 256, 8);
    pack_weights<<<1024, 256, 0, stream>>>(z2_W2, p_z2W2, 256, 8);
    pack_weights<<<1024, 256, 0, stream>>>(z2_U2, p_z2U2, 256, 8);
    pack_weights<<< 512, 256, 0, stream>>>(z1_W1, p_z1W1, 112, 4);
    pack_weights<<<1024, 256, 0, stream>>>(z1_U1, p_z1U1, 256, 8);
    pack_weights<<<1024, 256, 0, stream>>>(z1_W2, p_z1W2, 256, 8);
    pack_weights<<<1024, 256, 0, stream>>>(z1_U2, p_z1U2, 256, 8);

    EncArgs a;
    a.x = x; a.eps1 = eps1; a.eps2 = eps2;
    a.z2b1 = z2_b1; a.z2b2 = z2_b2; a.z1b1 = z1_b1; a.z1b2 = z1_b2;
    a.mu2W = mu2W; a.mu2b = mu2b; a.lv2W = lv2W; a.lv2b = lv2b;
    a.mu1W = mu1W; a.mu1b = mu1b; a.lv1W = lv1W; a.lv1b = lv1b;
    a.z2W1p = p_z2W1; a.z2U1p = p_z2U1; a.z2W2p = p_z2W2; a.z2U2p = p_z2U2;
    a.z1W1p = p_z1W1; a.z1U1p = p_z1U1; a.z1W2p = p_z1W2; a.z1U2p = p_z1U2;
    a.out = (float*)d_out;

    encoder_kernel<<<BATCH / 16, 256, 0, stream>>>(a);
}
